// bertCSRModel_7473243095239
// MI455X (gfx1250) — matrix operand order repaired and checked by reading, not compiled, not run on hardware
//
#include <hip/hip_runtime.h>
#include <hip/hip_bf16.h>
#include <stdint.h>

typedef __bf16 bf16_t;
typedef __attribute__((ext_vector_type(16))) __bf16 v16bf;
typedef __attribute__((ext_vector_type(8)))  float  v8f;
typedef __attribute__((ext_vector_type(4)))  unsigned int v4u;
typedef __attribute__((ext_vector_type(8)))  int v8i;
typedef __attribute__((ext_vector_type(4)))  int v4i;

#define NEV   4096
#define NP    32768
#define EDIM  768
#define CDIM  256
#define MLPH  1024
#define LSEQ  256
#define MSPAN 8

#define BM 128
#define BN 128
#define BK 32

// ---------------------------------------------------------------- utilities

__global__ void k_split_pairs(const int* __restrict__ pair_idx,
                              int* __restrict__ p0, int* __restrict__ p1) {
  int i = blockIdx.x * blockDim.x + threadIdx.x;
  if (i < NP) { p0[i] = pair_idx[2 * i]; p1[i] = pair_idx[2 * i + 1]; }
}

// e_emb[b, :] = masked mean of up to 8 tokens -> bf16
__global__ void k_event_emb(const float* __restrict__ sent,
                            const int* __restrict__ es,
                            const int* __restrict__ e0,
                            const int* __restrict__ e1,
                            bf16_t* __restrict__ out) {
  int b = blockIdx.x;
  int sid = es[b], st = e0[b], en = e1[b];
  float inv = 1.0f / (float)(en - st);
  const float* basep = sent + ((size_t)sid * LSEQ) * EDIM;
  for (int d = threadIdx.x; d < EDIM; d += 256) {
    float s = 0.f;
#pragma unroll
    for (int j = 0; j < MSPAN; ++j) {
      int pos = st + j;
      if (pos < en) s += basep[(size_t)pos * EDIM + d];
    }
    out[(size_t)b * EDIM + d] = (bf16_t)(s * inv);
  }
}

// transpose+convert weights: in fp32 [K][N] -> out bf16 [N][K]
__global__ void k_wT(const float* __restrict__ in, bf16_t* __restrict__ out,
                     int K, int N) {
  int gid = blockIdx.x * blockDim.x + threadIdx.x;
  if (gid < K * N) {
    int k = gid / N, n = gid % N;
    out[(size_t)n * K + k] = (bf16_t)in[gid];
  }
}

// diff_b[p, c] = bf16(x[p0[p], c] - x[p1[p], c])
__global__ void k_diff(const float* __restrict__ x,
                       const int* __restrict__ p0, const int* __restrict__ p1,
                       bf16_t* __restrict__ diffb) {
  int gid = blockIdx.x * blockDim.x + threadIdx.x;
  int p = gid >> 8, c = gid & 255;
  float v = x[(size_t)p0[p] * CDIM + c] - x[(size_t)p1[p] * CDIM + c];
  diffb[gid] = (bf16_t)v;
}

// ------------------------------------------------------- TDM 2D tile -> LDS
// Issue a Tensor Data Mover load of a 2D tile (tile_k bf16 elems contiguous
// per row, tile_rows rows, row pitch = stride_elems) into LDS at lds_off.
// D# layout per CDNA5 ISA ch.8 (group0: count/lds/global/type=2;
// group1: data_size=1(2B), dims/strides; groups 2-3 zero for 2D).
__device__ inline void tdm_load_2d(unsigned lds_off, const void* gaddr,
                                   unsigned tile_k, unsigned tile_rows,
                                   unsigned long long stride_elems) {
  unsigned long long ga = (unsigned long long)(uintptr_t)gaddr;
  v4u g0;
  g0.x = 0x1u;                                        // count=1, no gather
  g0.y = lds_off;                                     // lds_addr [63:32]
  g0.z = (unsigned)(ga & 0xffffffffu);                // global_addr lo
  g0.w = (unsigned)((ga >> 32) & 0x01ffffffu) | (2u << 30);  // hi25 | type=2
  unsigned td0 = tile_k;                              // tensor_dim0
  unsigned td1 = tile_rows;                           // tensor_dim1
  unsigned long long s0 = stride_elems;               // tensor_dim0_stride
  v8i g1;
  g1[0] = (int)(1u << 16);                            // data_size=1 (2 bytes)
  g1[1] = (int)((td0 & 0xffffu) << 16);               // td0[15:0] @ bits 63:48
  g1[2] = (int)(((td0 >> 16) & 0xffffu) | ((td1 & 0xffffu) << 16));
  g1[3] = (int)(((td1 >> 16) & 0xffffu) | ((tile_k & 0xffffu) << 16)); // tile_dim0
  g1[4] = (int)(tile_rows & 0xffffu);                 // tile_dim1; tile_dim2=0
  g1[5] = (int)(s0 & 0xffffffffu);                    // stride0 lo32
  g1[6] = (int)((s0 >> 32) & 0xffffu);                // stride0 hi16; stride1=0
  g1[7] = 0;
  v4i g2 = {0, 0, 0, 0};
  v4i g3 = {0, 0, 0, 0};
#if defined(__clang_major__) && __clang_major__ >= 23
  v8i g4 = {0, 0, 0, 0, 0, 0, 0, 0};
  __builtin_amdgcn_tensor_load_to_lds(g0, g1, g2, g3, g4, 0);
#else
  __builtin_amdgcn_tensor_load_to_lds(g0, g1, g2, g3, 0);
#endif
}

// ---------------------------------------------------------------- WMMA GEMM
// C[M,N] = A_bf16 @ Bt_bf16^T (+bias) (+addsrc) (relu?) -> Cf fp32 / Cb bf16
// Bt is [N][ldb] (pre-transposed weights, fed to LDS via TDM). Optional
// row-gather with K split for A: rows from g0 while k < kSplit, else g1.
__global__ __launch_bounds__(256)
void k_gemm(const bf16_t* __restrict__ A, int lda,
            const bf16_t* __restrict__ Bt, int ldb,
            const float* __restrict__ bias,
            const float* __restrict__ addsrc,
            float* __restrict__ Cf, bf16_t* __restrict__ Cb,
            const int* __restrict__ g0, const int* __restrict__ g1,
            int kSplit, int N, int K, int relu) {
  __shared__ __align__(16) bf16_t As[2][BM * BK];
  __shared__ __align__(16) bf16_t Bs[2][BN * BK];

  const int tid  = threadIdx.x;
  const int wave = tid >> 5;
  const int lane = tid & 31;
  const int mBase = blockIdx.y * BM;
  const int nBase = blockIdx.x * BN;
  const int nK = K / BK;
  const bool tdmA = (g0 == nullptr);

  v8f acc[8] = {};

  auto issue_tiles = [&](int buf, int kt) {
    const int k0 = kt * BK;
    if (!tdmA) {                           // gathered A: manual copy
#pragma unroll
      for (int j = 0; j < 2; ++j) {        // A: 128 x 32 bf16, 512 x 16B
        int c = j * 256 + tid;
        int row = c >> 2;
        int koff = (c & 3) * 8;
        int gr, gk;
        if (k0 < kSplit) { gr = g0[mBase + row]; gk = k0; }
        else             { gr = g1[mBase + row]; gk = k0 - kSplit; }
        uint4 v = *reinterpret_cast<const uint4*>(A + (size_t)gr * lda + gk + koff);
        *reinterpret_cast<uint4*>(&As[buf][row * BK + koff]) = v;
      }
    }
    if (wave == 0) {                       // one wave drives the TDM
      tdm_load_2d((unsigned)(uintptr_t)&Bs[buf][0],
                  Bt + (size_t)nBase * ldb + k0, BK, BN,
                  (unsigned long long)ldb);
      if (tdmA)
        tdm_load_2d((unsigned)(uintptr_t)&As[buf][0],
                    A + (size_t)mBase * lda + k0, BK, BM,
                    (unsigned long long)lda);
    }
  };

  issue_tiles(0, 0);
  const int mn  = lane & 15;               // row (A) / col (B) within tile
  const int kbA = (lane >> 4) * 8;         // A K-half base
  const int kbB = (lane >> 4) * 8;         // B K-half base

  union F { v16bf v; uint4 u[2]; };

  for (int kt = 0; kt < nK; ++kt) {
    if (wave == 0) __builtin_amdgcn_s_wait_tensorcnt(0);
    __syncthreads();
    if (kt + 1 < nK) issue_tiles((kt + 1) & 1, kt + 1);
    const int buf = kt & 1;

    F af;
    const bf16_t* ap = &As[buf][(wave * 16 + mn) * BK];
    af.u[0] = *reinterpret_cast<const uint4*>(ap + kbA);       // K = kb..kb+7
    af.u[1] = *reinterpret_cast<const uint4*>(ap + 16 + kbA);  // K = 16+kb..

    const bf16_t* bp0 = &Bs[buf][mn * BK + kbB];
    F bcur, bnext;
    bcur.u[0] = *reinterpret_cast<const uint4*>(bp0);
    bcur.u[1] = *reinterpret_cast<const uint4*>(bp0 + 16);
#pragma unroll
    for (int nt = 0; nt < 8; ++nt) {       // pipeline next B frag over WMMA
      if (nt < 7) {
        const bf16_t* bp = bp0 + (nt + 1) * 16 * BK;
        bnext.u[0] = *reinterpret_cast<const uint4*>(bp);
        bnext.u[1] = *reinterpret_cast<const uint4*>(bp + 16);
      }
      acc[nt] = __builtin_amdgcn_wmma_f32_16x16x32_bf16(
          false, af.v, false, bcur.v, (short)0, acc[nt], false, false);
      if (nt < 7) bcur = bnext;
    }
  }

  const int rowHalf = (lane >> 4) * 8;
  const int colIn = lane & 15;
#pragma unroll
  for (int nt = 0; nt < 8; ++nt) {
    int col = nBase + nt * 16 + colIn;
    float bv = bias ? bias[col] : 0.0f;
#pragma unroll
    for (int i = 0; i < 8; ++i) {
      int r = mBase + wave * 16 + rowHalf + i;
      float v = acc[nt][i] + bv;
      if (addsrc) v += addsrc[(size_t)r * N + col];
      if (relu) v = v > 0.f ? v : 0.f;
      if (Cf) Cf[(size_t)r * N + col] = v;
      if (Cb) Cb[(size_t)r * N + col] = (bf16_t)v;
    }
  }
}

// ---------------------------------------------------------------- pred / loss

// one wave per pair: pred = hidden[p] @ W2 + b2 ; masks + per-block loss partial
__global__ __launch_bounds__(256)
void k_pred_loss(const float* __restrict__ hidden,
                 const float* __restrict__ W2, const float* __restrict__ b2,
                 const int* __restrict__ target,
                 float* __restrict__ pred_out,        // [NP,3]
                 unsigned char* __restrict__ m1, unsigned char* __restrict__ m2,
                 float* __restrict__ loss_part) {
  __shared__ float sh_loss[8];
  int wave = threadIdx.x >> 5, lane = threadIdx.x & 31;
  int p = blockIdx.x * 8 + wave;
  const float* hp = hidden + (size_t)p * MLPH;
  float a0 = 0.f, a1 = 0.f, a2 = 0.f;
#pragma unroll 4
  for (int j = 0; j < MLPH / 32; ++j) {
    int k = lane + j * 32;
    float hv = hp[k];
    const float* w = W2 + k * 3;
    a0 += hv * w[0]; a1 += hv * w[1]; a2 += hv * w[2];
  }
#pragma unroll
  for (int off = 16; off > 0; off >>= 1) {
    a0 += __shfl_down(a0, off);
    a1 += __shfl_down(a1, off);
    a2 += __shfl_down(a2, off);
  }
  if (lane == 0) {
    float p0v = a0 + b2[0], p1v = a1 + b2[1], p2v = a2 + b2[2];
    pred_out[p * 3 + 0] = p0v; pred_out[p * 3 + 1] = p1v; pred_out[p * 3 + 2] = p2v;
    float mx = fmaxf(p0v, fmaxf(p1v, p2v));
    float se = expf(p0v - mx) + expf(p1v - mx) + expf(p2v - mx);
    int tgt = target[p];
    float pt = tgt == 0 ? p0v : (tgt == 1 ? p1v : p2v);
    sh_loss[wave] = -(pt - mx - logf(se));
    int arg = 0; float best = p0v;
    if (p1v > best) { best = p1v; arg = 1; }
    if (p2v > best) { best = p2v; arg = 2; }
    bool conf = (1.0f / se) > 0.5f;               // softmax max = 1/se
    m1[p] = (conf && arg == 1) ? 1 : 0;
    m2[p] = (conf && arg == 2) ? 1 : 0;
  }
  __syncthreads();
  if (threadIdx.x == 0) {
    float s = 0.f;
#pragma unroll
    for (int w = 0; w < 8; ++w) s += sh_loss[w];
    loss_part[blockIdx.x] = s;
  }
}

__global__ void k_loss_reduce(const float* __restrict__ part, float* __restrict__ out,
                              float scale, int first) {
  __shared__ float sh[256];
  float s = 0.f;
  for (int j = threadIdx.x; j < NP / 8; j += 256) s += part[j];
  sh[threadIdx.x] = s;
  __syncthreads();
  for (int off = 128; off > 0; off >>= 1) {
    if (threadIdx.x < off) sh[threadIdx.x] += sh[threadIdx.x + off];
    __syncthreads();
  }
  if (threadIdx.x == 0) {
    float v = scale * sh[0] / (float)NP;
    out[0] = first ? v : (out[0] + v);
  }
}

// ---------------------------------------------------------------- attention

// per-node head scores: [n][16] = {src_i h0..3, dst_i, src_x, dst_x}
__global__ void k_scores(const float* __restrict__ hbuf,
                         const float* __restrict__ asi, const float* __restrict__ adi,
                         const float* __restrict__ asx, const float* __restrict__ adx,
                         float* __restrict__ sv) {
  int gid = blockIdx.x * blockDim.x + threadIdx.x;
  if (gid >= NEV * 16) return;
  int n = gid >> 4, q = gid & 15;
  int sel = q >> 2, hh = q & 3;
  const float* att = sel == 0 ? asi : sel == 1 ? adi : sel == 2 ? asx : adx;
  const float* hp = hbuf + (size_t)n * CDIM + hh * 64;
  const float* ap = att + hh * 64;
  float s = 0.f;
#pragma unroll 8
  for (int d = 0; d < 64; ++d) s += hp[d] * ap[d];
  sv[gid] = s;
}

__global__ void k_fill(float* __restrict__ p, float v, int n) {
  int i = blockIdx.x * blockDim.x + threadIdx.x;
  if (i < n) p[i] = v;
}

__device__ inline void edge_decode(int e, int mode,
                                   const int* __restrict__ p0, const int* __restrict__ p1,
                                   const unsigned char* __restrict__ m1,
                                   const unsigned char* __restrict__ m2,
                                   const int* __restrict__ rel,
                                   int& s, int& d, bool& mask) {
  if (e < 2 * NP) {
    int pe = e < NP ? e : e - NP;
    int ss = e < NP ? p0[pe] : p1[pe];
    int dd = e < NP ? p1[pe] : p0[pe];
    bool mm = (e < NP ? m1[pe] : m2[pe]) != 0;
    bool r0 = rel[pe] == 0;
    s = ss; d = dd;
    mask = (mode == 0) ? (mm && r0) : (mm && !r0);
  } else {                                 // self loops (intra only)
    s = d = e - 2 * NP; mask = true;
  }
}

__device__ inline float leaky(float a) { return a > 0.f ? a : 0.2f * a; }

__device__ inline void atomicMaxF(float* addr, float val) {
  int* ia = (int*)addr;
  int old = __float_as_int(*addr);
  while (__int_as_float(old) < val) {
    int assumed = old;
    old = atomicCAS(ia, assumed, __float_as_int(val));
    if (old == assumed) break;
  }
}

__global__ void k_att_max(int mode, int E,
                          const int* p0, const int* p1,
                          const unsigned char* m1, const unsigned char* m2,
                          const int* rel, const float* __restrict__ sv,
                          float* __restrict__ amax) {
  int gid = blockIdx.x * blockDim.x + threadIdx.x;
  if (gid >= E * 4) return;
  int e = gid >> 2, h = gid & 3;
  int s, d; bool mask;
  edge_decode(e, mode, p0, p1, m1, m2, rel, s, d, mask);
  if (!mask) return;
  int offS = mode == 0 ? 0 : 8, offD = mode == 0 ? 4 : 12;
  float a = leaky(sv[s * 16 + offS + h] + sv[d * 16 + offD + h]);
  atomicMaxF(&amax[d * 4 + h], a);
}

__global__ void k_att_exp(int mode, int E,
                          const int* p0, const int* p1,
                          const unsigned char* m1, const unsigned char* m2,
                          const int* rel, const float* __restrict__ sv,
                          const float* __restrict__ amax,
                          float* __restrict__ denom, float* __restrict__ exb) {
  int gid = blockIdx.x * blockDim.x + threadIdx.x;
  if (gid >= E * 4) return;
  int e = gid >> 2, h = gid & 3;
  int s, d; bool mask;
  edge_decode(e, mode, p0, p1, m1, m2, rel, s, d, mask);
  float ex = 0.f;
  if (mask) {
    int offS = mode == 0 ? 0 : 8, offD = mode == 0 ? 4 : 12;
    float a = leaky(sv[s * 16 + offS + h] + sv[d * 16 + offD + h]);
    ex = expf(a - amax[d * 4 + h]);
    atomicAdd(&denom[d * 4 + h], ex);
  }
  exb[gid] = ex;
}

// one block per edge; 256 channels
__global__ __launch_bounds__(256)
void k_att_agg(int mode,
               const int* p0, const int* p1,
               const unsigned char* m1, const unsigned char* m2,
               const int* rel, const float* __restrict__ exb,
               const float* __restrict__ denom, const float* __restrict__ hbuf,
               float* __restrict__ outb) {
  int e = blockIdx.x;
  int s, d; bool mask;
  edge_decode(e, mode, p0, p1, m1, m2, rel, s, d, mask);
  if (!mask) return;
  int c = threadIdx.x, h = c >> 6;
  float ex = exb[e * 4 + h];
  if (ex == 0.f) return;
  float al = ex / (denom[d * 4 + h] + 1e-9f);
  atomicAdd(&outb[(size_t)d * CDIM + c], al * hbuf[(size_t)s * CDIM + c]);
}

__global__ void k_combine(const float* __restrict__ oi, const float* __restrict__ ox,
                          float* __restrict__ x, bf16_t* __restrict__ xb) {
  int gid = blockIdx.x * blockDim.x + threadIdx.x;
  float v = 0.5f * oi[gid] + 0.5f * ox[gid];
  x[gid] = v;
  xb[gid] = (bf16_t)v;
}

// ---------------------------------------------------------------- launch

extern "C" void kernel_launch(void* const* d_in, const int* in_sizes, int n_in,
                              void* d_out, int out_size, void* d_ws, size_t ws_size,
                              hipStream_t stream) {
  const float* sent_emb   = (const float*)d_in[0];
  const float* proj_in_W  = (const float*)d_in[1];
  const float* proj_in_b  = (const float*)d_in[2];
  const float* proj_W     = (const float*)d_in[3];
  const float* proj_b     = (const float*)d_in[4];
  const float* asi        = (const float*)d_in[5];
  const float* adi        = (const float*)d_in[6];
  const float* asx        = (const float*)d_in[7];
  const float* adx        = (const float*)d_in[8];
  const float* mlp_W1     = (const float*)d_in[9];
  const float* mlp_b1     = (const float*)d_in[10];
  const float* mlp_W2     = (const float*)d_in[11];
  const float* mlp_b2     = (const float*)d_in[12];
  const int*   event_sent = (const int*)d_in[13];
  const int*   event_start= (const int*)d_in[14];
  const int*   event_end  = (const int*)d_in[15];
  const int*   pair_idx   = (const int*)d_in[16];
  const int*   rel_type   = (const int*)d_in[17];
  const int*   target     = (const int*)d_in[18];
  float* out = (float*)d_out;

  char* ws = (char*)d_ws;
  size_t off = 0;
  auto alloc = [&](size_t bytes) -> char* {
    char* p = ws + off;
    off = (off + bytes + 255) & ~(size_t)255;
    return p;
  };
  int*    p0      = (int*)alloc((size_t)NP * 4);
  int*    p1      = (int*)alloc((size_t)NP * 4);
  bf16_t* e_emb_b = (bf16_t*)alloc((size_t)NEV * EDIM * 2);
  bf16_t* Wint    = (bf16_t*)alloc((size_t)EDIM * CDIM * 2);      // [256][768]
  bf16_t* Wpt     = (bf16_t*)alloc((size_t)CDIM * CDIM * 2);      // [256][256]
  bf16_t* W1t     = (bf16_t*)alloc((size_t)MLPH * 1792 * 2);      // [1024][1792]
  float*  x       = (float*)alloc((size_t)NEV * CDIM * 4);
  bf16_t* xb      = (bf16_t*)alloc((size_t)NEV * CDIM * 2);
  float*  base    = (float*)alloc((size_t)NP * MLPH * 4);
  float*  hidden  = (float*)alloc((size_t)NP * MLPH * 4);
  bf16_t* diffb   = (bf16_t*)alloc((size_t)NP * CDIM * 2);
  unsigned char* m1 = (unsigned char*)alloc(NP);
  unsigned char* m2 = (unsigned char*)alloc(NP);
  float*  hbuf    = (float*)alloc((size_t)NEV * CDIM * 4);
  float*  sv      = (float*)alloc((size_t)NEV * 16 * 4);
  float*  amax    = (float*)alloc((size_t)2 * NEV * 4 * 4);   // intra | inter
  float*  denom   = (float*)alloc((size_t)2 * NEV * 4 * 4);
  float*  outs    = (float*)alloc((size_t)2 * NEV * CDIM * 4);
  float*  ex_i    = (float*)alloc((size_t)(2 * NP + NEV) * 4 * 4);
  float*  ex_x    = (float*)alloc((size_t)(2 * NP) * 4 * 4);
  float*  lpart   = (float*)alloc((size_t)(NP / 8) * 4);
  (void)ws_size; (void)in_sizes; (void)n_in; (void)out_size;

  float* amax_i = amax,  * amax_x = amax + NEV * 4;
  float* den_i  = denom, * den_x  = denom + NEV * 4;
  float* out_i  = outs,  * out_x  = outs + (size_t)NEV * CDIM;

  const int EI = 2 * NP + NEV, EX = 2 * NP;

  // ---- setup
  k_split_pairs<<<(NP + 255) / 256, 256, 0, stream>>>(pair_idx, p0, p1);
  k_event_emb<<<NEV, 256, 0, stream>>>(sent_emb, event_sent, event_start, event_end, e_emb_b);
  k_wT<<<(EDIM * CDIM + 255) / 256, 256, 0, stream>>>(proj_in_W, Wint, EDIM, CDIM);
  k_wT<<<(CDIM * CDIM + 255) / 256, 256, 0, stream>>>(proj_W, Wpt, CDIM, CDIM);
  k_wT<<<(1792 * MLPH + 255) / 256, 256, 0, stream>>>(mlp_W1, W1t, 1792, MLPH);

  // base = e_emb[p0] @ W1[:768] + e_emb[p1] @ W1[768:1536] + b1   (once)
  k_gemm<<<dim3(MLPH / BN, NP / BM), 256, 0, stream>>>(
      e_emb_b, EDIM, W1t, 1792, mlp_b1, nullptr, base, nullptr,
      p0, p1, EDIM, MLPH, 2 * EDIM, 0);

  // x0 = e_emb @ proj_in_W + b
  k_gemm<<<dim3(CDIM / BN, NEV / BM), 256, 0, stream>>>(
      e_emb_b, EDIM, Wint, EDIM, proj_in_b, nullptr, x, xb,
      nullptr, nullptr, 0, CDIM, EDIM, 0);

  for (int it = 0; it < 3; ++it) {
    // diff -> bf16
    k_diff<<<NP * CDIM / 256, 256, 0, stream>>>(x, p0, p1, diffb);

    // hidden = relu(base + diff @ W1[1536:1792])
    k_gemm<<<dim3(MLPH / BN, NP / BM), 256, 0, stream>>>(
        diffb, CDIM, W1t + 1536, 1792, nullptr, base, hidden, nullptr,
        nullptr, nullptr, 0, MLPH, CDIM, 1);

    // pred + masks + loss
    k_pred_loss<<<NP / 8, 256, 0, stream>>>(hidden, mlp_W2, mlp_b2, target,
                                            out + 1, m1, m2, lpart);
    k_loss_reduce<<<1, 256, 0, stream>>>(lpart, out, 1.0f / (float)(it + 1), it == 0);

    // h = x @ proj_W + b
    k_gemm<<<dim3(CDIM / BN, NEV / BM), 256, 0, stream>>>(
        xb, CDIM, Wpt, CDIM, proj_b, nullptr, hbuf, nullptr,
        nullptr, nullptr, 0, CDIM, CDIM, 0);

    // attention
    k_scores<<<(NEV * 16 + 255) / 256, 256, 0, stream>>>(hbuf, asi, adi, asx, adx, sv);
    k_fill<<<(2 * NEV * 4 + 255) / 256, 256, 0, stream>>>(amax, -1e9f, 2 * NEV * 4);
    hipMemsetAsync(denom, 0, (size_t)2 * NEV * 4 * 4, stream);
    hipMemsetAsync(outs, 0, (size_t)2 * NEV * CDIM * 4, stream);

    k_att_max<<<(EI * 4 + 255) / 256, 256, 0, stream>>>(0, EI, p0, p1, m1, m2, rel_type, sv, amax_i);
    k_att_max<<<(EX * 4 + 255) / 256, 256, 0, stream>>>(1, EX, p0, p1, m1, m2, rel_type, sv, amax_x);
    k_att_exp<<<(EI * 4 + 255) / 256, 256, 0, stream>>>(0, EI, p0, p1, m1, m2, rel_type, sv, amax_i, den_i, ex_i);
    k_att_exp<<<(EX * 4 + 255) / 256, 256, 0, stream>>>(1, EX, p0, p1, m1, m2, rel_type, sv, amax_x, den_x, ex_x);
    k_att_agg<<<EI, 256, 0, stream>>>(0, p0, p1, m1, m2, rel_type, ex_i, den_i, hbuf, out_i);
    k_att_agg<<<EX, 256, 0, stream>>>(1, p0, p1, m1, m2, rel_type, ex_x, den_x, hbuf, out_x);

    // x = 0.5*intra + 0.5*inter
    k_combine<<<NEV * CDIM / 256, 256, 0, stream>>>(out_i, out_x, x, xb);
  }
}
